// Discriminator_26396869001791
// MI455X (gfx1250) — matrix operand order repaired and checked by reading, not compiled, not run on hardware
//
#include <hip/hip_runtime.h>
#include <hip/hip_bf16.h>
#include <math.h>

// ---------------- types for CDNA5 WMMA ----------------
typedef __attribute__((ext_vector_type(16))) __bf16 v16bf;
typedef __attribute__((ext_vector_type(8)))  float  v8f;

#define DFEA 128
#define BN_EPS 1e-5f

// ---------------------------------------------------------------------------
// Convert W_ft [128x128] and W_gnn [3x128x128] (f32, row-major) into
// bf16 TRANSPOSED matrices: Wt[m][col*128 + row] so each output-column's
// K-vector is contiguous (32B-aligned v16bf loads for WMMA B operand).
// ---------------------------------------------------------------------------
__global__ __launch_bounds__(256) void k_cvt_weights(const float* __restrict__ Wft,
                                                     const float* __restrict__ Wgnn,
                                                     __bf16* __restrict__ Wt) {
    int idx = blockIdx.x * 256 + threadIdx.x;      // 0 .. 4*128*128-1
    int m = idx >> 14;                             // matrix 0..3
    int r = (idx >> 7) & 127;                      // source row (K index)
    int c = idx & 127;                             // source col (N index)
    const float* src = (m == 0) ? Wft : (Wgnn + (size_t)(m - 1) * 16384);
    Wt[(size_t)m * 16384 + c * 128 + r] = (__bf16)src[r * 128 + c];
}

__global__ __launch_bounds__(256) void k_zero(float* __restrict__ p, int n) {
    int i = blockIdx.x * 256 + threadIdx.x;
    if (i < n) p[i] = 0.0f;
}

// ---------------------------------------------------------------------------
// Edge scatter: agg[dst] += h[src].  One feature per lane, 2 edges per block.
// Coalesced 128-lane gather + 128 coalesced f32 atomics per edge.
// ---------------------------------------------------------------------------
__global__ __launch_bounds__(256) void k_scatter_add(const float* __restrict__ h,
                                                     const int* __restrict__ ei,
                                                     float* __restrict__ agg, int E) {
    int e = blockIdx.x * 2 + (threadIdx.x >> 7);
    int t = threadIdx.x & 127;
    if (e < E) {
        int s = ei[e];
        int d = ei[E + e];
        atomicAdd(&agg[(size_t)d * DFEA + t], h[(size_t)s * DFEA + t]);
    }
}

// ---------------------------------------------------------------------------
// WMMA GEMM: Out[N,128] = (A1 (+A2)) [N,128] @ Wt^T + bias  (optional ReLU).
// Block = 64 threads (2 waves); wave owns 4 column tiles of a 16-row strip
// -> 4 independent accumulator chains.  Software-pipelined: all raw loads
// for K-step kb+1 are issued BEFORE the 4 WMMAs of step kb; the f32->bf16
// converts for kb+1 sit after the WMMAs and co-execute with the XDL pipe.
// K = 128 as 4 x (4 x v_wmma_f32_16x16x32_bf16).  EXEC all-ones (N%16==0).
// ---------------------------------------------------------------------------
template <bool ADD2, bool RELU>
__global__ __launch_bounds__(64) void k_gemm128(const float* __restrict__ A1,
                                                const float* __restrict__ A2,
                                                const __bf16* __restrict__ Wt,
                                                const float* __restrict__ bias,
                                                float* __restrict__ Out) {
    const int wave = threadIdx.x >> 5;        // 0..1 -> 64-column half
    const int lane = threadIdx.x & 31;
    const int half = lane >> 4;               // lane group 0/1
    const int ln   = lane & 15;
    const int rowBase = blockIdx.x * 16;
    const int colBase = wave * 64;            // this wave's 4 tiles: colBase + 16t

    // C init: bias broadcast per tile column.
    v8f c0, c1, c2, c3;
    {
        float b0 = bias[colBase + ln];
        float b1 = bias[colBase + 16 + ln];
        float b2 = bias[colBase + 32 + ln];
        float b3 = bias[colBase + 48 + ln];
#pragma unroll
        for (int i = 0; i < 8; ++i) { c0[i] = b0; c1[i] = b1; c2[i] = b2; c3[i] = b3; }
    }

    const float4* a1 = (const float4*)(A1 + (size_t)(rowBase + ln) * DFEA);
    const float4* a2 = (const float4*)(A2 + (size_t)(rowBase + ln) * DFEA);
    const __bf16* w0 = Wt + (size_t)(colBase + ln) * DFEA;
    const __bf16* w1 = Wt + (size_t)(colBase + 16 + ln) * DFEA;
    const __bf16* w2 = Wt + (size_t)(colBase + 32 + ln) * DFEA;
    const __bf16* w3 = Wt + (size_t)(colBase + 48 + ln) * DFEA;

    // ---- pipeline helpers (all compile-time unrolled) ----
    auto loadA = [&](int kb, float4* p, float4* q) {
        const int k0 = kb * 32;
        const int i0 = (k0 + half * 16) >> 2;       // elems 0..15 -> K = k0 + 16*half + e
        p[0] = a1[i0]; p[1] = a1[i0 + 1]; p[2] = a1[i0 + 2]; p[3] = a1[i0 + 3];
        if (ADD2) { q[0] = a2[i0]; q[1] = a2[i0 + 1]; q[2] = a2[i0 + 2]; q[3] = a2[i0 + 3]; }
    };
    auto loadB = [&](int kb, v16bf* b) {
        const int koff = kb * 32 + half * 16;       // elem e -> K = koff + e
        b[0] = *(const v16bf*)(w0 + koff);
        b[1] = *(const v16bf*)(w1 + koff);
        b[2] = *(const v16bf*)(w2 + koff);
        b[3] = *(const v16bf*)(w3 + koff);
    };
    auto mkA = [&](const float4* p, const float4* q) -> v16bf {
        float4 p0 = p[0], p1 = p[1], p2 = p[2], p3 = p[3];
        if (ADD2) {
            p0.x += q[0].x; p0.y += q[0].y; p0.z += q[0].z; p0.w += q[0].w;
            p1.x += q[1].x; p1.y += q[1].y; p1.z += q[1].z; p1.w += q[1].w;
            p2.x += q[2].x; p2.y += q[2].y; p2.z += q[2].z; p2.w += q[2].w;
            p3.x += q[3].x; p3.y += q[3].y; p3.z += q[3].z; p3.w += q[3].w;
        }
        v16bf a;
        a[0]  = (__bf16)p0.x; a[1]  = (__bf16)p0.y; a[2]  = (__bf16)p0.z; a[3]  = (__bf16)p0.w;
        a[4]  = (__bf16)p1.x; a[5]  = (__bf16)p1.y; a[6]  = (__bf16)p1.z; a[7]  = (__bf16)p1.w;
        a[8]  = (__bf16)p2.x; a[9]  = (__bf16)p2.y; a[10] = (__bf16)p2.z; a[11] = (__bf16)p2.w;
        a[12] = (__bf16)p3.x; a[13] = (__bf16)p3.y; a[14] = (__bf16)p3.z; a[15] = (__bf16)p3.w;
        return a;
    };

    // ---- prologue: stage K-step 0 ----
    float4 pc[4], qc[4];
    v16bf  bc[4];
    loadA(0, pc, qc);
    loadB(0, bc);
    v16bf ac = mkA(pc, qc);

#pragma unroll
    for (int kb = 0; kb < 4; ++kb) {
        float4 pn[4], qn[4];
        v16bf  bn[4];
        if (kb < 3) {                 // 1) issue ALL raw loads for kb+1 first
            loadA(kb + 1, pn, qn);
            loadB(kb + 1, bn);
        }
        // 2) 4 independent WMMAs for kb (operands already resident)
        c0 = __builtin_amdgcn_wmma_f32_16x16x32_bf16(false, ac, false, bc[0], (short)0, c0, false, false);
        c1 = __builtin_amdgcn_wmma_f32_16x16x32_bf16(false, ac, false, bc[1], (short)0, c1, false, false);
        c2 = __builtin_amdgcn_wmma_f32_16x16x32_bf16(false, ac, false, bc[2], (short)0, c2, false, false);
        c3 = __builtin_amdgcn_wmma_f32_16x16x32_bf16(false, ac, false, bc[3], (short)0, c3, false, false);
        if (kb < 3) {                 // 3) converts co-execute with XDL pipe
            ac = mkA(pn, qn);
            bc[0] = bn[0]; bc[1] = bn[1]; bc[2] = bn[2]; bc[3] = bn[3];
        }
    }

    // C/D layout: VGPR r -> row (r + 8*half), column = tile col.
#pragma unroll
    for (int r = 0; r < 8; ++r) {
        float* orow = Out + (size_t)(rowBase + r + 8 * half) * DFEA + colBase + ln;
        float v0 = c0[r], v1 = c1[r], v2 = c2[r], v3 = c3[r];
        if (RELU) {
            v0 = fmaxf(v0, 0.0f); v1 = fmaxf(v1, 0.0f);
            v2 = fmaxf(v2, 0.0f); v3 = fmaxf(v3, 0.0f);
        }
        orow[0]  = v0;
        orow[16] = v1;
        orow[32] = v2;
        orow[48] = v3;
    }
}

// ---------------------------------------------------------------------------
// BatchNorm pass 1: per-column sum / sumsq (grid-stride, register accumulate,
// one atomicAdd pair per thread at the end).  sums[0..127]=Σx, [128..255]=Σx²
// ---------------------------------------------------------------------------
__global__ __launch_bounds__(256) void k_bn_stats(const float* __restrict__ X,
                                                  float* __restrict__ sums, int N) {
    int c  = threadIdx.x & 127;
    int r0 = blockIdx.x * 2 + (threadIdx.x >> 7);
    int stride = gridDim.x * 2;
    float s = 0.0f, ss = 0.0f;
    for (int r = r0; r < N; r += stride) {
        float v = X[(size_t)r * DFEA + c];
        s += v;
        ss += v * v;
    }
    atomicAdd(&sums[c], s);
    atomicAdd(&sums[128 + c], ss);
}

// BatchNorm pass 2: fold into per-column scale/shift.
__global__ __launch_bounds__(128) void k_bn_final(const float* __restrict__ sums,
                                                  const float* __restrict__ gamma,
                                                  const float* __restrict__ beta,
                                                  float* __restrict__ params,
                                                  float invN) {
    int c = threadIdx.x;
    float mu  = sums[c] * invN;
    float var = sums[128 + c] * invN - mu * mu;   // biased variance
    float sc  = gamma[c] * rsqrtf(var + BN_EPS);
    params[c]       = sc;
    params[128 + c] = beta[c] - mu * sc;
}

// BatchNorm apply + ReLU; optionally mirror into the embedding output.
template <bool WRITE_EMB>
__global__ __launch_bounds__(256) void k_bn_apply(const float* __restrict__ X,
                                                  const float* __restrict__ params,
                                                  float* __restrict__ H,
                                                  float* __restrict__ Emb,
                                                  int total) {
    int i = blockIdx.x * 256 + threadIdx.x;
    if (i >= total) return;
    int c = i & 127;
    float v = fmaxf(X[i] * params[c] + params[128 + c], 0.0f);
    H[i] = v;
    if (WRITE_EMB) Emb[i] = v;
}

// ---------------------------------------------------------------------------
// Head: out[r] = sigmoid(dot(emb[r], W_mlp) + b).  One wave per row,
// float4 per lane + shuffle reduction (warpSize == 32 on gfx1250).
// ---------------------------------------------------------------------------
__global__ __launch_bounds__(256) void k_head(const float* __restrict__ Emb,
                                              const float* __restrict__ Wm,
                                              const float* __restrict__ bm,
                                              float* __restrict__ out, int N) {
    int wave = threadIdx.x >> 5;
    int lane = threadIdx.x & 31;
    int row  = blockIdx.x * 8 + wave;
    if (row >= N) return;
    const float4 hv = ((const float4*)(Emb + (size_t)row * DFEA))[lane];
    const float4 wv = ((const float4*)Wm)[lane];
    float s = hv.x * wv.x + hv.y * wv.y + hv.z * wv.z + hv.w * wv.w;
#pragma unroll
    for (int o = 16; o > 0; o >>= 1) s += __shfl_down(s, o, 32);
    if (lane == 0) out[row] = 1.0f / (1.0f + expf(-(s + bm[0])));
}

// ---------------------------------------------------------------------------
extern "C" void kernel_launch(void* const* d_in, const int* in_sizes, int n_in,
                              void* d_out, int out_size, void* d_ws, size_t ws_size,
                              hipStream_t stream) {
    const float* x      = (const float*)d_in[0];
    const int*   ei     = (const int*)  d_in[1];   // [2,E]
    const float* W_ft   = (const float*)d_in[2];
    const float* b_ft   = (const float*)d_in[3];
    const float* W_gnn  = (const float*)d_in[4];   // [3,128,128]
    const float* b_gnn  = (const float*)d_in[5];   // [3,128]
    const float* gamma  = (const float*)d_in[6];
    const float* beta   = (const float*)d_in[7];
    const float* W_mlp  = (const float*)d_in[8];   // [128,1]
    const float* b_mlp  = (const float*)d_in[9];

    const int N = in_sizes[0] / DFEA;              // 50000 (divisible by 16)
    const int E = in_sizes[1] / 2;                 // 800000
    const size_t NH = (size_t)N * DFEA;

    float* ws     = (float*)d_ws;
    float* h      = ws;                 // [N,128]
    float* tmp    = ws + NH;            // [N,128]
    float* agg    = ws + 2 * NH;        // [N,128]  (contiguous with sums)
    float* sums   = ws + 3 * NH;        // [256]
    float* params = ws + 3 * NH + 256;  // [256]
    __bf16* Wt    = (__bf16*)(ws + 3 * NH + 512);  // 4 x 128x128 bf16 (transposed)

    float* out_p = (float*)d_out;       // [N]
    float* emb_p = out_p + N;           // [N,128]

    const int gemmBlocks    = N / 16;   // 16 rows per block
    const int ewBlocks      = (N * DFEA + 255) / 256;
    const int zeroN         = (int)NH + 256;
    const int zeroBlocks    = (zeroN + 255) / 256;
    const int scatterBlocks = (E + 1) / 2;

    // 0) weights -> bf16 transposed
    k_cvt_weights<<<256, 256, 0, stream>>>(W_ft, W_gnn, Wt);

    // 1) h = relu(x @ W_ft + b_ft)
    k_gemm128<false, true><<<gemmBlocks, 64, 0, stream>>>(x, nullptr, Wt, b_ft, h);

    // 2) three GIN layers
    for (int l = 0; l < 3; ++l) {
        k_zero<<<zeroBlocks, 256, 0, stream>>>(agg, zeroN);   // agg + sums
        k_scatter_add<<<scatterBlocks, 256, 0, stream>>>(h, ei, agg, E);
        k_gemm128<true, false><<<gemmBlocks, 64, 0, stream>>>(
            h, agg, Wt + (size_t)(l + 1) * 16384, b_gnn + l * DFEA, tmp);
        k_bn_stats<<<512, 256, 0, stream>>>(tmp, sums, N);
        k_bn_final<<<1, 128, 0, stream>>>(sums, gamma + l * DFEA, beta + l * DFEA,
                                          params, 1.0f / (float)N);
        if (l == 2) {
            k_bn_apply<true><<<ewBlocks, 256, 0, stream>>>(tmp, params, h, emb_p, (int)NH);
        } else {
            k_bn_apply<false><<<ewBlocks, 256, 0, stream>>>(tmp, params, h, nullptr, (int)NH);
        }
    }

    // 3) head
    k_head<<<(N + 7) / 8, 256, 0, stream>>>(emb_p, W_mlp, b_mlp, out_p, N);
}
